// ParallelizedCPMKernel_60215441490498
// MI455X (gfx1250) — compile-verified, matrix operand order checked by reading, not run on hardware
//
#include <hip/hip_runtime.h>
#include <stdint.h>

// ---------------------------------------------------------------------------
// Cellular Potts Model Metropolis step for MI455X (gfx1250).
// Bandwidth-bound problem (~100 MB traffic @ 23.3 TB/s HBM, L2-resident):
//  - bulk cpm copy done by the Tensor Data Mover (tensor_load_to_lds /
//    tensor_store_from_lds, TENSORcnt-tracked) instead of VALU loops
//  - per-wave energy reduction done with v_wmma_f32_16x16x32_f16
//    (32 lane partials -> one matrix op -> one cross-half shuffle)
// ---------------------------------------------------------------------------

#define HH     2048
#define WW     2048
#define HWTOT  (HH * WW)
#define KPROP  4096
#define TDM_ROWS 4   // rows per TDM tile: 4*2048*4B = 32 KB LDS

typedef __attribute__((ext_vector_type(16))) _Float16 v16h;
typedef __attribute__((ext_vector_type(8)))  float    v8f;
typedef __attribute__((ext_vector_type(4)))  unsigned v4u;
typedef __attribute__((ext_vector_type(8)))  int      v8i;
typedef __attribute__((ext_vector_type(4)))  int      v4i;

// --- counter-based PRNG (PCG output function) ------------------------------
__device__ __forceinline__ unsigned pcg(unsigned& state) {
  state = state * 747796405u + 2891336453u;
  unsigned w = ((state >> ((state >> 28u) + 4u)) ^ state) * 277803737u;
  return (w >> 22u) ^ w;
}

// --- wave32 sum via WMMA: D = A x ones, A holds lane partials --------------
// lane m<16 puts x into A[m][0], lane m+16 into A[m][8]; with B=ones the row
// sums of D give x[m]+x[m+16]; summing the 8 D VGPRs per lane gives the sum
// of rows 0..7 (lanes<16) / 8..15 (lanes>=16); one xor-16 shuffle finishes.
__device__ __forceinline__ float wave_sum_wmma(float x) {
  v16h a = {};
  a[0] = (_Float16)x;
  v16h b;
#pragma unroll
  for (int i = 0; i < 16; ++i) b[i] = (_Float16)1.0f;
  v8f c = {};
  c = __builtin_amdgcn_wmma_f32_16x16x32_f16(false, a, false, b,
                                             (short)0, c, false, false);
  float r = c[0] + c[1] + c[2] + c[3] + c[4] + c[5] + c[6] + c[7];
  r += __shfl_xor(r, 16, 32);
  return r;
}

// --- ws accumulator init ----------------------------------------------------
__global__ void cpm_zero_ws(float* wsf) {
  if (threadIdx.x < 8) wsf[threadIdx.x] = 0.0f;
}

// --- bulk cpm copy via Tensor Data Mover -----------------------------------
// Treat cpm [2,2048,2048] f32 as 4096 rows x 2048 cols; each 1-wave block
// DMAs a 4-row tile global->LDS->global.
__global__ __launch_bounds__(32) void cpm_tdm_copy(const float* __restrict__ src,
                                                   float* __restrict__ dst) {
  __shared__ float tile[TDM_ROWS * WW];
  // low 32 bits of the flat shared address are the LDS byte offset
  unsigned lds = (unsigned)(size_t)(void*)tile;

  unsigned long long off = (unsigned long long)blockIdx.x *
                           (unsigned long long)(TDM_ROWS * WW * 4);
  unsigned long long gs = (unsigned long long)(size_t)src + off;
  unsigned long long gd = (unsigned long long)(size_t)dst + off;

  // D# group1: data_size=4B(code2), tensor 2048 x TDM_ROWS, tile 2048 x
  // TDM_ROWS, dim0 stride 2048 elements, no padding/iterate/multicast.
  v8i g1 = { (int)(2u << 16),                 // data_size=2 (4B), wg_mask=0
             (int)((unsigned)WW << 16),       // tensor_dim0 lo16 @ bits 63:48
             (int)((unsigned)TDM_ROWS << 16), // dim0 hi16 | tensor_dim1 lo16
             (int)((unsigned)WW << 16),       // dim1 hi16 | tile_dim0
             (int)TDM_ROWS,                   // tile_dim1 | tile_dim2(0)
             (int)WW,                         // tensor_dim0_stride lo32
             0, 0 };                          // stride0 hi | stride1 (unused)
  v4i gz  = { 0, 0, 0, 0 };                   // groups 2/3: 2-D tensor
  v8i gz8 = { 0, 0, 0, 0, 0, 0, 0, 0 };       // unused trailing group

  // D# group0: count=1, lds_addr, 57-bit global addr, type=2 ("image")
  v4u g0l = { 1u, lds, (unsigned)(gs & 0xFFFFFFFFull),
              (unsigned)((gs >> 32) & 0x1FFFFFFull) | 0x80000000u };
  v4u g0s = { 1u, lds, (unsigned)(gd & 0xFFFFFFFFull),
              (unsigned)((gd >> 32) & 0x1FFFFFFull) | 0x80000000u };

  __builtin_amdgcn_tensor_load_to_lds(g0l, g1, gz, gz, gz8, 0);
  __builtin_amdgcn_s_wait_tensorcnt(0);
  __builtin_amdgcn_tensor_store_from_lds(g0s, g1, gz, gz, gz8, 0);
  __builtin_amdgcn_s_wait_tensorcnt(0);
}

// --- phase 1: generate K proposals, evaluate Metropolis criterion ----------
__global__ __launch_bounds__(128) void cpm_propose(
    const float* __restrict__ cpm, const float* __restrict__ bmask,
    const float* __restrict__ temp, const int* __restrict__ seedp,
    int* __restrict__ tgt, int* __restrict__ acc,
    float* __restrict__ pvid, float* __restrict__ pvt,
    float* __restrict__ wsf) {
  int k = blockIdx.x * 128 + threadIdx.x;
  const float* ids = cpm;
  const float* ts  = cpm + HWTOT;

  unsigned s = (unsigned)seedp[0] * 0x9E3779B9u + (unsigned)k * 0x85EBCA6Bu
             + 0x6C62272Eu;

  // boundary-site sampling (random ids => nearly every site is boundary, so
  // rejection sampling converges almost immediately)
  int flat = 0;
#pragma unroll 1
  for (int it = 0; it < 16; ++it) {
    flat = (int)(pcg(s) & (HWTOT - 1));
    if (bmask[flat] > 0.5f) break;
  }
  int sx = flat >> 11, sy = flat & (WW - 1);

  const int dx[4] = { -1, 1, 0, 0 };
  const int dy[4] = { 0, 0, -1, 1 };
  int d  = (int)(pcg(s) & 3u);
  int nx = (sx + dx[d]) & (HH - 1);
  int ny = (sy + dy[d]) & (WW - 1);
  int nflat = nx * WW + ny;

  float vid = ids[flat],  vt = ts[flat];    // source spin copied to neighbor
  float cid = ids[nflat], ct = ts[nflat];   // neighbor's current spin

  // local adhesion energy at the neighbor, old vs proposed
  float eo = 0.0f, en = 0.0f;
#pragma unroll
  for (int i = 0; i < 4; ++i) {
    int ax = (nx + dx[i]) & (HH - 1);
    int ay = (ny + dy[i]) & (WW - 1);
    float nid = ids[ax * WW + ay], nt = ts[ax * WW + ay];
    float dco = nt - ct, dcn = nt - vt;
    eo += (nid != cid) ? (1.0f + 0.5f * dco * dco) : 0.0f;
    en += (nid != vid) ? (1.0f + 0.5f * dcn * dcn) : 0.0f;
  }

  float dE = (en - eo) / temp[0];
  float u  = (float)(pcg(s) >> 8) * (1.0f / 16777216.0f);
  int a = (u < expf(-dE)) ? 1 : 0;

  tgt[k] = nflat; acc[k] = a; pvid[k] = vid; pvt[k] = vt;
  atomicAdd(&wsf[1], dE);                       // sum of deltas
  if (a) { atomicAdd(&wsf[2], dE); atomicAdd(&wsf[3], 1.0f); }
}

// --- phase 2: deterministic commit (last proposal index wins per target) ---
__global__ __launch_bounds__(128) void cpm_commit(
    const int* __restrict__ tgt, const int* __restrict__ acc,
    const float* __restrict__ pvid, const float* __restrict__ pvt,
    float* __restrict__ out_cpm) {
  int k = blockIdx.x * 128 + threadIdx.x;
  if (!acc[k]) return;                 // rejected: neighbor keeps old value
  int t = tgt[k];
  for (int j = k + 1; j < KPROP; ++j)
    if (tgt[j] == t) return;           // a later proposal owns this site
  out_cpm[t]         = pvid[k];
  out_cpm[HWTOT + t] = pvt[k];
}

// --- fused boundary mask + adhesion energy over updated grid ---------------
__global__ __launch_bounds__(256) void cpm_boundary_energy(
    const float* __restrict__ cpm, float* __restrict__ bout,
    float* __restrict__ wsf) {
  const float* ids = cpm;
  const float* ts  = cpm + HWTOT;
  int idx = blockIdx.x * 256 + threadIdx.x;
  int y  = idx & (WW - 1);
  int x  = idx >> 11;
  int xp = (x + 1) & (HH - 1), xm = (x - 1) & (HH - 1);
  int yp = (y + 1) & (WW - 1), ym = (y - 1) & (WW - 1);

  float id  = ids[idx];
  float idR = ids[xp * WW + y];
  float idL = ids[xm * WW + y];
  float idD = ids[x * WW + yp];
  float idU = ids[x * WW + ym];
  float t   = ts[idx];
  float tR  = ts[xp * WW + y];
  float tD  = ts[x * WW + yp];

  float dR = t - tR, dD = t - tD;
  float e = ((id != idR) ? (1.0f + 0.5f * dR * dR) : 0.0f)
          + ((id != idD) ? (1.0f + 0.5f * dD * dD) : 0.0f);

  bout[idx] = ((id != idR) | (id != idL) | (id != idD) | (id != idU))
                  ? 1.0f : 0.0f;

  float tot = wave_sum_wmma(e);        // EXEC all-ones: exact grid, no branch
  if ((threadIdx.x & 31) == 0) atomicAdd(&wsf[0], tot);
}

// --- scalars ----------------------------------------------------------------
__global__ void cpm_finalize(const float* __restrict__ wsf,
                             const float* __restrict__ orig,
                             const float* __restrict__ temp,
                             float* __restrict__ out) {
  float T = temp[0];
  float energy = wsf[0] / T;
  out[2 * HWTOT] = energy;                       // energy (already /T)
  float invK = 1.0f / (float)KPROP;
  float* sc = out + (2 * HWTOT + 1 + HWTOT);
  sc[0] = wsf[1] * invK;                         // deltas.mean()
  sc[1] = wsf[2] * invK;                         // (deltas*accepts).sum()/K
  sc[2] = (energy - orig[0]) * invK;             // delta_true / K
  sc[3] = wsf[3] * invK;                         // accept rate
  sc[4] = wsf[3];                                // accepts.sum()
}

extern "C" void kernel_launch(void* const* d_in, const int* in_sizes, int n_in,
                              void* d_out, int out_size, void* d_ws,
                              size_t ws_size, hipStream_t stream) {
  const float* cpm   = (const float*)d_in[0];   // [2,2048,2048]
  const float* orig  = (const float*)d_in[1];   // scalar
  const float* bmask = (const float*)d_in[2];   // [2048,2048]
  const float* temp  = (const float*)d_in[3];   // scalar
  const int*   seed  = (const int*)d_in[4];     // scalar

  float* out  = (float*)d_out;
  float* wsf  = (float*)d_ws;                   // [0..7] accumulators
  int*   tgt  = (int*)(wsf + 8);                // [K]
  int*   acc  = tgt + KPROP;                    // [K]
  float* pvid = (float*)(acc + KPROP);          // [K]
  float* pvt  = pvid + KPROP;                   // [K]

  cpm_zero_ws<<<1, 32, 0, stream>>>(wsf);
  // 2 planes * 2048 rows / TDM_ROWS rows-per-tile
  cpm_tdm_copy<<<(2 * HH) / TDM_ROWS, 32, 0, stream>>>(cpm, out);
  cpm_propose<<<KPROP / 128, 128, 0, stream>>>(cpm, bmask, temp, seed,
                                               tgt, acc, pvid, pvt, wsf);
  cpm_commit<<<KPROP / 128, 128, 0, stream>>>(tgt, acc, pvid, pvt, out);
  cpm_boundary_energy<<<HWTOT / 256, 256, 0, stream>>>(
      out, out + (2 * HWTOT + 1), wsf);
  cpm_finalize<<<1, 1, 0, stream>>>(wsf, orig, temp, out);
}
